// ResonanceLayer_89266600280536
// MI455X (gfx1250) — matrix operand order repaired and checked by reading, not compiled, not run on hardware
//
#include <hip/hip_runtime.h>
#include <hip/hip_bf16.h>
#include <math.h>

#define B_     32
#define N_     256
#define T_OBS_ 50
#define T_S_   25
#define D_     128
#define DH_    256
#define P_     8
#define TWO_PI_ 6.283185307179586f
#define EPS_    1e-4f

typedef __attribute__((ext_vector_type(16))) _Float16 v16h;
typedef __attribute__((ext_vector_type(8)))  _Float16 v8h;
typedef __attribute__((ext_vector_type(8)))  float    v8f;

// Workspace element counts
#define FN_RED_ELEMS   (800 * D_)        // f32
#define FN_OUT_ELEMS   (800 * 64)        // f32
#define W1_ELEMS       (D_ * DH_)        // 32768
#define W2_ELEMS       (DH_ * DH_)       // 65536
#define W3_ELEMS       (DH_ * 64)        // 16384

// ---------------------------------------------------------------------------
// Kernel 0: convert MLP weights to f16 once (deterministic, re-done per call).
// ---------------------------------------------------------------------------
__launch_bounds__(256)
__global__ void k0_cvt(const float* __restrict__ W1,
                       const float* __restrict__ W2,
                       const float* __restrict__ W3,
                       _Float16* __restrict__ W1h,
                       _Float16* __restrict__ W2h,
                       _Float16* __restrict__ W3h) {
    const int i = blockIdx.x * blockDim.x + threadIdx.x;
    if (i < W1_ELEMS) {
        W1h[i] = (_Float16)W1[i];
    } else if (i < W1_ELEMS + W2_ELEMS) {
        const int j = i - W1_ELEMS;
        W2h[j] = (_Float16)W2[j];
    } else if (i < W1_ELEMS + W2_ELEMS + W3_ELEMS) {
        const int j = i - (W1_ELEMS + W2_ELEMS);
        W3h[j] = (_Float16)W3[j];
    }
}

// ---------------------------------------------------------------------------
// Kernel 1: fn_red[b,t,d] = max_n( f_ego[b,t,d] * f_nei[b,n,t,d] )
// One block per (b,t). 8 waves split the n range (wave w: n = w, w+8, ...),
// each iteration reads one fully contiguous 512 B row. Deterministic LDS
// tree-max merges the 8 partials (max is order-invariant anyway).
// ---------------------------------------------------------------------------
__launch_bounds__(256)
__global__ void k1_maxmul(const float* __restrict__ f_ego,
                          const float* __restrict__ f_nei,
                          float* __restrict__ fn_red) {
    __shared__ float4 sm[256];

    const int bt  = blockIdx.x;            // 0..799
    const int b   = bt / T_S_;
    const int t   = bt % T_S_;
    const int d4  = threadIdx.x & 31;      // 32 float4 = 128 floats
    const int w   = threadIdx.x >> 5;      // wave id 0..7 (n-group)

    const float4 ego = reinterpret_cast<const float4*>(f_ego)[(b * T_S_ + t) * 32 + d4];
    float4 m = make_float4(-INFINITY, -INFINITY, -INFINITY, -INFINITY);

    const float4* base = reinterpret_cast<const float4*>(f_nei)
                       + ((size_t)(b * N_) * T_S_ + t) * 32 + d4;
    const size_t nstride = (size_t)T_S_ * 32;   // one neighbor slab, float4 units

    for (int n = w; n < N_; n += 8) {
        if (n + 16 < N_)
            __builtin_prefetch(base + (size_t)(n + 16) * nstride, 0, 1);
        float4 v = base[(size_t)n * nstride];
        m.x = fmaxf(m.x, ego.x * v.x);
        m.y = fmaxf(m.y, ego.y * v.y);
        m.z = fmaxf(m.z, ego.z * v.z);
        m.w = fmaxf(m.w, ego.w * v.w);
    }

    sm[w * 32 + d4] = m;
    __syncthreads();
    #pragma unroll
    for (int s = 4; s >= 1; s >>= 1) {
        if (w < s) {
            float4 a = sm[w * 32 + d4];
            float4 c = sm[(w + s) * 32 + d4];
            a.x = fmaxf(a.x, c.x); a.y = fmaxf(a.y, c.y);
            a.z = fmaxf(a.z, c.z); a.w = fmaxf(a.w, c.w);
            sm[w * 32 + d4] = a;
        }
        __syncthreads();
    }
    if (w == 0)
        reinterpret_cast<float4*>(fn_red)[(b * T_S_ + t) * 32 + d4] = sm[d4];
}

// ---------------------------------------------------------------------------
// WMMA fragment loaders (v_wmma_f32_16x16x32_f16 operand layouts, ISA 7.12.2)
// A (16x32 f16): lane%16 = M row; lane/16 = K-octet group:
//   halves 0..7 -> K = 8g + j ; halves 8..15 -> K = 16 + 8g + (j-8)
// B (32x16 f16): lane%16 = N column; lane/16 = K-octet group, same K order as A.
// All offsets are multiples of 8 halves -> aligned 16 B vector loads.
// ---------------------------------------------------------------------------
__device__ __forceinline__ v16h load_a_lds(const _Float16* A, int ldk, int k0, int lane) {
    const int m = lane & 15;
    const int g = lane >> 4;
    const _Float16* row = A + m * ldk + k0 + 8 * g;
    v8h lo = *(const v8h*)(row);
    v8h hi = *(const v8h*)(row + 16);
    v16h a;
    #pragma unroll
    for (int j = 0; j < 8; ++j) { a[j] = lo[j]; a[j + 8] = hi[j]; }
    return a;
}

__device__ __forceinline__ v16h load_b_h(const _Float16* Wh, int ldn, int k0, int n0, int lane) {
    const int g = lane >> 4;
    const _Float16* col = Wh + (size_t)(k0 + 8 * g) * ldn + n0 + (lane & 15);
    v16h b;
    #pragma unroll
    for (int j = 0; j < 8; ++j) { b[j] = col[(size_t)j * ldn]; b[j + 8] = col[(size_t)(16 + j) * ldn]; }
    return b;
}

// ---------------------------------------------------------------------------
// Kernel 2: 3-layer MLP on (800 x 128) via WMMA. One workgroup per 16-row
// M-tile (800 = 50*16), 16 waves: each wave owns one 16-col N-tile.
// ---------------------------------------------------------------------------
__launch_bounds__(512)
__global__ void k2_mlp(const float* __restrict__ X,        // (800,128) fn_red
                       const _Float16* __restrict__ W1h, const float* __restrict__ b1,
                       const _Float16* __restrict__ W2h, const float* __restrict__ b2,
                       const _Float16* __restrict__ W3h, const float* __restrict__ b3,
                       float* __restrict__ Y) {            // (800,64)
    __shared__ _Float16 Xh[16 * D_];     // 4 KB
    __shared__ _Float16 H1[16 * DH_];    // 8 KB
    __shared__ _Float16 H2[16 * DH_];    // 8 KB

    const int tid  = threadIdx.x;
    const int lane = tid & 31;
    const int w    = tid >> 5;           // wave id 0..15
    const int g    = lane >> 4;
    const int l15  = lane & 15;
    const int m0   = blockIdx.x * 16;

    for (int i = tid; i < 16 * D_; i += 512)
        Xh[i] = (_Float16)X[(size_t)(m0 + (i >> 7)) * D_ + (i & 127)];
    __syncthreads();

    // ---- Layer 1: (16x128) @ (128x256) ----
    {
        const int n0 = w * 16;
        v8f c;
        const float bias = b1[n0 + l15];
        #pragma unroll
        for (int r = 0; r < 8; ++r) c[r] = bias;
        #pragma unroll
        for (int kc = 0; kc < 4; ++kc) {
            v16h a  = load_a_lds(Xh, D_, kc * 32, lane);
            v16h bb = load_b_h(W1h, DH_, kc * 32, n0, lane);
            c = __builtin_amdgcn_wmma_f32_16x16x32_f16(false, a, false, bb,
                                                       (short)0, c, false, false);
        }
        #pragma unroll
        for (int r = 0; r < 8; ++r)
            H1[(r + 8 * g) * DH_ + n0 + l15] = (_Float16)fmaxf(c[r], 0.0f);
    }
    __syncthreads();

    // ---- Layer 2: (16x256) @ (256x256) ----
    {
        const int n0 = w * 16;
        v8f c;
        const float bias = b2[n0 + l15];
        #pragma unroll
        for (int r = 0; r < 8; ++r) c[r] = bias;
        #pragma unroll
        for (int kc = 0; kc < 8; ++kc) {
            v16h a  = load_a_lds(H1, DH_, kc * 32, lane);
            v16h bb = load_b_h(W2h, DH_, kc * 32, n0, lane);
            c = __builtin_amdgcn_wmma_f32_16x16x32_f16(false, a, false, bb,
                                                       (short)0, c, false, false);
        }
        #pragma unroll
        for (int r = 0; r < 8; ++r)
            H2[(r + 8 * g) * DH_ + n0 + l15] = (_Float16)fmaxf(c[r], 0.0f);
    }
    __syncthreads();

    // ---- Layer 3: (16x256) @ (256x64) ---- (4 waves active)
    if (w < 4) {
        const int n0 = w * 16;
        v8f c;
        const float bias = b3[n0 + l15];
        #pragma unroll
        for (int r = 0; r < 8; ++r) c[r] = bias;
        #pragma unroll
        for (int kc = 0; kc < 8; ++kc) {
            v16h a  = load_a_lds(H2, DH_, kc * 32, lane);
            v16h bb = load_b_h(W3h, 64, kc * 32, n0, lane);
            c = __builtin_amdgcn_wmma_f32_16x16x32_f16(false, a, false, bb,
                                                       (short)0, c, false, false);
        }
        #pragma unroll
        for (int r = 0; r < 8; ++r)
            Y[(size_t)(m0 + r + 8 * g) * 64 + n0 + l15] = fmaxf(c[r], 0.0f);
    }
}

// ---------------------------------------------------------------------------
// Kernel 3: geometry, deterministic 8-bucket binning, output assembly.
// One block per (b,t); thread n handles neighbor n.
// ---------------------------------------------------------------------------
__launch_bounds__(256)
__global__ void k3_geo(const float* __restrict__ x_ego,
                       const float* __restrict__ x_nei,
                       const float* __restrict__ Y,        // (800,64)
                       const float* __restrict__ Wce,      // (2,64)
                       const float* __restrict__ bce,      // (64,)
                       float* __restrict__ out) {          // (B,25,8,128)
    __shared__ float dist_sh[N_];
    __shared__ float ang_sh[N_];
    __shared__ int   pidx_sh[N_];
    __shared__ float dmean[P_], amean[P_], scale[P_];

    const int bt   = blockIdx.x;
    const int b    = bt / T_S_;
    const int t    = bt % T_S_;
    const int tobs = 2 * t;                 // ::2 slice of T_OBS
    const int n    = threadIdx.x;

    const float xe0 = x_ego[(b * T_OBS_ + tobs) * 2 + 0];
    const float xe1 = x_ego[(b * T_OBS_ + tobs) * 2 + 1];
    const float xn0 = x_nei[((size_t)(b * N_ + n) * T_OBS_ + tobs) * 2 + 0];
    const float xn1 = x_nei[((size_t)(b * N_ + n) * T_OBS_ + tobs) * 2 + 1];

    const float dx = xn0 - xe0;
    const float dy = xn1 - xe1;
    const float dist = sqrtf(dx * dx + dy * dy);
    float ang = atan2f(dx, dy);             // reference: arctan2(p[...,0], p[...,1])
    if (ang < 0.0f) ang += TWO_PI_;
    const int p = (int)(ang * ((float)P_ / TWO_PI_));
    const bool mask = ((fabsf(dx) + fabsf(dy)) != 0.0f) && (dist > 0.005f);

    pidx_sh[n] = mask ? p : -1;             // p==8 (exact 2*pi) matches no bucket, as in ref
    dist_sh[n] = dist;
    ang_sh[n]  = ang;
    __syncthreads();

    if (n < P_) {                           // deterministic serial bucket sums
        int cnt = 0; float sd = 0.0f, sa = 0.0f;
        for (int i = 0; i < N_; ++i) {
            if (pidx_sh[i] == n) { cnt++; sd += dist_sh[i]; sa += ang_sh[i]; }
        }
        const float np = (float)cnt + EPS_;
        dmean[n] = sd / np;
        amean[n] = sa / np;
        scale[n] = (float)cnt / np;
    }
    __syncthreads();

    const float* y = Y   + (size_t)bt * 64;
    float*       o = out + (size_t)bt * (P_ * 128);
    for (int idx = n; idx < P_ * 128; idx += 256) {
        const int p2 = idx >> 7;
        const int d  = idx & 127;
        float v;
        if (d < 64) {
            v = y[d] * scale[p2];
        } else {
            const int j = d - 64;
            v = fmaxf(dmean[p2] * Wce[j] + amean[p2] * Wce[64 + j] + bce[j], 0.0f);
        }
        o[idx] = v;
    }
}

// ---------------------------------------------------------------------------
extern "C" void kernel_launch(void* const* d_in, const int* in_sizes, int n_in,
                              void* d_out, int out_size, void* d_ws, size_t ws_size,
                              hipStream_t stream) {
    (void)in_sizes; (void)n_in; (void)out_size; (void)ws_size;
    const float* x_ego = (const float*)d_in[0];
    const float* x_nei = (const float*)d_in[1];
    const float* f_ego = (const float*)d_in[2];
    const float* f_nei = (const float*)d_in[3];
    const float* W1    = (const float*)d_in[4];
    const float* b1    = (const float*)d_in[5];
    const float* W2    = (const float*)d_in[6];
    const float* b2    = (const float*)d_in[7];
    const float* W3    = (const float*)d_in[8];
    const float* b3    = (const float*)d_in[9];
    const float* Wce   = (const float*)d_in[10];
    const float* bce   = (const float*)d_in[11];
    float* out = (float*)d_out;

    // Workspace layout (all 16B-aligned: hipMalloc base + multiples of 16)
    float*    fn_red = (float*)d_ws;                          // 800*128 f32
    float*    fn_out = fn_red + FN_RED_ELEMS;                 // 800*64  f32
    _Float16* W1h    = (_Float16*)(fn_out + FN_OUT_ELEMS);    // 32768 f16
    _Float16* W2h    = W1h + W1_ELEMS;                        // 65536 f16
    _Float16* W3h    = W2h + W2_ELEMS;                        // 16384 f16

    const int cvt_total = W1_ELEMS + W2_ELEMS + W3_ELEMS;     // 114688
    k0_cvt   <<<(cvt_total + 255) / 256, 256, 0, stream>>>(W1, W2, W3, W1h, W2h, W3h);
    k1_maxmul<<<800, 256, 0, stream>>>(f_ego, f_nei, fn_red);
    k2_mlp   <<<50, 512, 0, stream>>>(fn_red, W1h, b1, W2h, b2, W3h, b3, fn_out);
    k3_geo   <<<800, 256, 0, stream>>>(x_ego, x_nei, fn_out, Wce, bce, out);
}
